// EquivariantStructureDenoisingModule_39127152066621
// MI455X (gfx1250) — matrix operand order repaired and checked by reading, not compiled, not run on hardware
//
#include <hip/hip_runtime.h>
#include <hip/hip_bf16.h>
#include <math.h>

#define N_NODES 20000
#define DIM     128
#define NH      8
#define NP      8
#define NE      320000
#define NL      4
#define EPS_C   1e-4f

typedef __attribute__((ext_vector_type(16))) _Float16 v16h;
typedef __attribute__((ext_vector_type(8)))  _Float16 v8h;
typedef __attribute__((ext_vector_type(8)))  float    v8f;

__device__ __forceinline__ float gelu_exact(float x) {
  return 0.5f * x * (1.0f + erff(x * 0.7071067811865475f));
}
// monotonic float<->uint mapping for atomicMax on f32
__device__ __forceinline__ unsigned f2ord(float f) {
  unsigned u = __float_as_uint(f);
  return (u & 0x80000000u) ? ~u : (u | 0x80000000u);
}
__device__ __forceinline__ float ord2f(unsigned u) {
  return (u & 0x80000000u) ? __uint_as_float(u & 0x7FFFFFFFu) : __uint_as_float(~u);
}

// ---------------------------------------------------------------------------
// Tiled WMMA GEMM on pre-converted f16 operands.
//   C[M,Nc] (+)= act( A[M,K] @ B[K,Nc] + bias )
// A:  f16 row-major, lda elements/row, rows optionally gathered (A-row=gather[m])
// Bt: f16 *transposed* weights, Bt[n, k], ldbt elements per n-row (>= K slice)
// C:  f32 (read when ACC, written when STORE_F32); Ch: optional f16 output.
// flags: bit0 ACC (v += C), bit1 GELU, bit2 store f32 C.
// Each wave owns a 16(M) x 64(N) tile; 4 accumulators reuse one A fragment.
// M % 16 == 0, K % 32 == 0, Nc % 64 == 0 at every call site.
// ---------------------------------------------------------------------------
__global__ void __launch_bounds__(128)
gemm_wmma_f16(const _Float16* __restrict__ A, const int* __restrict__ gather, int lda,
              const _Float16* __restrict__ Bt, int ldbt,
              float* __restrict__ C, _Float16* __restrict__ Ch, int ldc,
              int M, int K,
              const float* __restrict__ bias, int flags)
{
  const int wave = threadIdx.x >> 5;
  const int lane = threadIdx.x & 31;
  const int m0 = (blockIdx.x * 4 + wave) * 16;
  if (m0 >= M) return;
  const int n0   = blockIdx.y * 64;
  const int half = lane >> 4;
  const int l15  = lane & 15;

  const int row  = m0 + l15;
  const int arow = gather ? gather[row] : row;
  const _Float16* Ar = A + (size_t)arow * lda;

  v8f acc[4] = {};

  for (int kb = 0; kb < K; kb += 32) {
    // A fragment 16x32 (ISA 7.12.2): lanes 0-15 K {0..7,16..23},
    // lanes 16-31 K {8..15,24..31}; contiguous f16 -> b128 loads.
    const _Float16* ap = Ar + kb + half * 8;
    v8h a0 = *(const v8h*)(ap);
    v8h a1 = *(const v8h*)(ap + 16);
    v16h af = __builtin_shufflevector(a0, a1, 0, 1, 2, 3, 4, 5, 6, 7,
                                      8, 9, 10, 11, 12, 13, 14, 15);
    const int kk = kb + half * 8;  // B: same K split as A
#pragma unroll
    for (int j = 0; j < 4; ++j) {
      const _Float16* bp = Bt + (size_t)(n0 + j * 16 + l15) * ldbt + kk;
      v8h b0 = *(const v8h*)(bp);
      v8h b1 = *(const v8h*)(bp + 16);
      v16h bf = __builtin_shufflevector(b0, b1, 0, 1, 2, 3, 4, 5, 6, 7,
                                        8, 9, 10, 11, 12, 13, 14, 15);
      acc[j] = __builtin_amdgcn_wmma_f32_16x16x32_f16(false, af, false, bf,
                                                      (short)0, acc[j], false, false);
    }
  }

  // C/D layout: VGPR r -> row m0 + r + 8*half, col = tile + (lane&15)
#pragma unroll
  for (int j = 0; j < 4; ++j) {
    const int col = n0 + j * 16 + l15;
#pragma unroll
    for (int r = 0; r < 8; ++r) {
      const int rowm = m0 + r + half * 8;
      float v = acc[j][r];
      if (bias) v += bias[col];
      const size_t o = (size_t)rowm * ldc + col;
      if (flags & 1) v += C[o];
      if (flags & 2) v = gelu_exact(v);
      if (flags & 4) C[o] = v;
      if (Ch) Ch[o] = (_Float16)v;
    }
  }
}

// ------------------------------ helpers ------------------------------------
__global__ void fill_f32(float* p, int n, float v) {
  int i = blockIdx.x * 256 + threadIdx.x;
  if (i < n) p[i] = v;
}
__global__ void fill_u32(unsigned* p, int n, unsigned v) {
  int i = blockIdx.x * 256 + threadIdx.x;
  if (i < n) p[i] = v;
}
// f32 -> f16 copy
__global__ void conv_h(const float* __restrict__ a, _Float16* __restrict__ b, int n) {
  int i = blockIdx.x * 256 + threadIdx.x;
  if (i < n) b[i] = (_Float16)a[i];
}
// weight transpose+convert: Bt[n,k] = (f16) W[k,n]
__global__ void wconv_t(const float* __restrict__ W, _Float16* __restrict__ Bt,
                        int K, int Nc) {
  int tid = blockIdx.x * 256 + threadIdx.x;
  if (tid >= K * Nc) return;
  int n = tid / K, k = tid - n * K;
  Bt[(size_t)n * K + k] = (_Float16)W[(size_t)k * Nc + n];
}

// st[n,0:128]=x[n,0,:]; st[n,128:256]=t[n,:]; sbuf = layer-entry s (both f16)
__global__ void build_st(const float* __restrict__ x, const float* __restrict__ t,
                         _Float16* __restrict__ st, _Float16* __restrict__ sbuf) {
  int tid = blockIdx.x * 256 + threadIdx.x;
  if (tid >= N_NODES * 256) return;
  int n = tid >> 8, c = tid & 255;
  float v;
  if (c < 128) { v = x[(size_t)n * 512 + c]; sbuf[(size_t)n * 128 + c] = (_Float16)v; }
  else         { v = t[(size_t)n * 128 + (c - 128)]; }
  st[tid] = (_Float16)v;
}

// vbuf[(n*3+i), c] = x[n, 1+i, c]  (contiguous 3N x 128, f16)
__global__ void pack_v(const float* __restrict__ x, _Float16* __restrict__ vb) {
  int tid = blockIdx.x * 256 + threadIdx.x;
  if (tid >= 3 * N_NODES * 128) return;
  int r = tid >> 7, c = tid & 127;
  int n = r / 3, i = r - n * 3;
  vb[tid] = (_Float16)x[(size_t)n * 512 + (1 + i) * 128 + c];
}

// logits[e,h] = q[dst]·k[src]/4 + (xe@We)[e,h] - softplus(gamma[h]) * dist
__global__ void edge_logits(const float* __restrict__ q, const float* __restrict__ k,
                            const float* __restrict__ pq, const float* __restrict__ pk,
                            const float* __restrict__ xe, const float* __restrict__ We,
                            const float* __restrict__ gamma,
                            const int* __restrict__ src, const int* __restrict__ dst,
                            float* __restrict__ logits) {
  int tid = blockIdx.x * 256 + threadIdx.x;
  if (tid >= NE * NH) return;
  int e = tid >> 3, h = tid & 7;
  int s = src[e], d = dst[e];
  const float* qr = q + (size_t)d * 128 + h * 16;
  const float* kr = k + (size_t)s * 128 + h * 16;
  float dot = 0.f;
#pragma unroll
  for (int i = 0; i < 16; ++i) dot += qr[i] * kr[i];
  dot *= 0.25f;                       // 1/sqrt(16)
  float dist = 0.f;
#pragma unroll
  for (int i = 0; i < 3; ++i) {
    const float* a = pq + ((size_t)d * 3 + i) * 64 + h * 8;
    const float* b = pk + ((size_t)s * 3 + i) * 64 + h * 8;
#pragma unroll
    for (int p = 0; p < 8; ++p) { float dd = a[p] - b[p]; dist += dd * dd; }
  }
  const float* xer = xe + (size_t)e * 128;
  float eb = 0.f;
  for (int c = 0; c < 128; ++c) eb += xer[c] * We[c * 8 + h];
  float g  = gamma[h];
  float sp = (g > 20.f) ? g : log1pf(expf(g));
  logits[tid] = dot + eb - sp * dist;
}

__global__ void seg_max(const float* __restrict__ logits, const int* __restrict__ dst,
                        unsigned* __restrict__ m) {
  int tid = blockIdx.x * 256 + threadIdx.x;
  if (tid >= NE * NH) return;
  int e = tid >> 3, h = tid & 7;
  atomicMax(&m[(size_t)dst[e] * 8 + h], f2ord(logits[tid]));
}

__global__ void edge_w(float* __restrict__ logits, const int* __restrict__ dst,
                       const unsigned* __restrict__ m, float* __restrict__ denom) {
  int tid = blockIdx.x * 256 + threadIdx.x;
  if (tid >= NE * NH) return;
  int e = tid >> 3, h = tid & 7;
  float mm = ord2f(m[(size_t)dst[e] * 8 + h]);
  float w = expf(logits[tid] - mm);
  logits[tid] = w;
  atomicAdd(&denom[(size_t)dst[e] * 8 + h], w);
}

__global__ void edge_agg(const float* __restrict__ w, const int* __restrict__ src,
                         const int* __restrict__ dst, const float* __restrict__ denom,
                         const float* __restrict__ val, const float* __restrict__ pv,
                         float* __restrict__ out_s, float* __restrict__ out_p) {
  int tid = blockIdx.x * 256 + threadIdx.x;
  if (tid >= NE * NH) return;
  int e = tid >> 3, h = tid & 7;
  int s = src[e], d = dst[e];
  float alpha = w[tid] / denom[(size_t)d * 8 + h];
  const float* vr = val + (size_t)s * 128 + h * 16;
  float* os = out_s + (size_t)d * 128 + h * 16;
#pragma unroll
  for (int i = 0; i < 16; ++i) atomicAdd(&os[i], alpha * vr[i]);
#pragma unroll
  for (int i = 0; i < 3; ++i) {
    const float* pr = pv + ((size_t)s * 3 + i) * 64 + h * 8;
    float* op = out_p + ((size_t)d * 3 + i) * 64 + h * 8;
#pragma unroll
    for (int p = 0; p < 8; ++p) atomicAdd(&op[p], alpha * pr[p]);
  }
}

// catb[n,0:128]=out_s; catb[n,128+j]=||out_p[n,:,j]||   (f16)
__global__ void build_cat(const float* __restrict__ out_s, const float* __restrict__ out_p,
                          _Float16* __restrict__ catb) {
  int tid = blockIdx.x * 256 + threadIdx.x;
  if (tid >= N_NODES * 192) return;
  int n = tid / 192, c = tid - n * 192;
  float v;
  if (c < 128) v = out_s[(size_t)n * 128 + c];
  else {
    int j = c - 128;
    float a = out_p[((size_t)n * 3 + 0) * 64 + j];
    float b = out_p[((size_t)n * 3 + 1) * 64 + j];
    float d = out_p[((size_t)n * 3 + 2) * 64 + j];
    v = sqrtf(a * a + b * b + d * d + EPS_C);
  }
  catb[tid] = (_Float16)v;
}

__global__ void x_update1(float* __restrict__ x, const float* __restrict__ s_out,
                          const float* __restrict__ v_out) {
  int tid = blockIdx.x * 256 + threadIdx.x;
  if (tid >= N_NODES * 128) return;
  int n = tid >> 7, c = tid & 127;
  x[(size_t)n * 512 + c] += s_out[tid];
#pragma unroll
  for (int i = 0; i < 3; ++i)
    x[(size_t)n * 512 + (1 + i) * 128 + c] += v_out[((size_t)n * 3 + i) * 128 + c];
}

// z[n] = [s2 | vloc(c-major,(c,j)) | vn]   (f16)
__global__ void build_z(const float* __restrict__ x, const float* __restrict__ R,
                        _Float16* __restrict__ z) {
  int tid = blockIdx.x * 256 + threadIdx.x;
  if (tid >= N_NODES * 128) return;
  int n = tid >> 7, c = tid & 127;
  const float* xn = x + (size_t)n * 512;
  float a0 = xn[128 + c], a1 = xn[256 + c], a2 = xn[384 + c];
  const float* Rn = R + (size_t)n * 9;
  _Float16* zn = z + (size_t)n * 640;
  zn[c] = (_Float16)xn[c];
#pragma unroll
  for (int j = 0; j < 3; ++j)
    zn[128 + c * 3 + j] = (_Float16)(a0 * Rn[j] + a1 * Rn[3 + j] + a2 * Rn[6 + j]);
  zn[512 + c] = (_Float16)sqrtf(a0 * a0 + a1 * a1 + a2 * a2 + EPS_C);
}

__global__ void x_update2(float* __restrict__ x, const float* __restrict__ R,
                          const float* __restrict__ h2) {
  int tid = blockIdx.x * 256 + threadIdx.x;
  if (tid >= N_NODES * 128) return;
  int n = tid >> 7, c = tid & 127;
  const float* hn = h2 + (size_t)n * 512;
  const float* Rn = R + (size_t)n * 9;
  x[(size_t)n * 512 + c] += hn[c];
  float b0 = hn[128 + c], b1 = hn[256 + c], b2 = hn[384 + c];
#pragma unroll
  for (int i = 0; i < 3; ++i)
    x[(size_t)n * 512 + (1 + i) * 128 + c] +=
        Rn[i * 3 + 0] * b0 + Rn[i * 3 + 1] * b1 + Rn[i * 3 + 2] * b2;
}

// ---------------------------------------------------------------------------
static inline unsigned nblk(size_t n) { return (unsigned)((n + 255) / 256); }

static inline void gemm(hipStream_t s, const _Float16* A, const int* g, int lda,
                        const _Float16* Bt, int ldbt, float* C, _Float16* Ch, int ldc,
                        int M, int K, int Nc, const float* bias, int flags) {
  dim3 grid((M + 63) / 64, Nc / 64);
  gemm_wmma_f16<<<grid, dim3(128), 0, s>>>(A, g, lda, Bt, ldbt, C, Ch, ldc, M, K,
                                           bias, flags);
}

extern "C" void kernel_launch(void* const* d_in, const int* in_sizes, int n_in,
                              void* d_out, int out_size, void* d_ws, size_t ws_size,
                              hipStream_t stream) {
  (void)in_sizes; (void)n_in; (void)out_size; (void)ws_size;
  const float* in_x  = (const float*)d_in[0];
  const float* in_t  = (const float*)d_in[1];
  const float* in_R  = (const float*)d_in[2];
  const float* in_xe = (const float*)d_in[3];
  const int*   ei    = (const int*)d_in[4];
  const float* Wq  = (const float*)d_in[5];
  const float* Wk  = (const float*)d_in[6];
  const float* Wv  = (const float*)d_in[7];
  const float* Wpq = (const float*)d_in[8];
  const float* Wpk = (const float*)d_in[9];
  const float* Wpv = (const float*)d_in[10];
  const float* We  = (const float*)d_in[11];
  const float* gam = (const float*)d_in[12];
  const float* Wos = (const float*)d_in[13];
  const float* Wov = (const float*)d_in[14];
  const float* We1 = (const float*)d_in[15];
  const float* We2 = (const float*)d_in[16];
  const float* lW1 = (const float*)d_in[17];
  const float* lb1 = (const float*)d_in[18];
  const float* lW2 = (const float*)d_in[19];
  const float* lb2 = (const float*)d_in[20];
  const int* src = ei;
  const int* dst = ei + NE;

  // outputs double as working state: x (N,4,128) then x_edge (E,128)
  float* x  = (float*)d_out;
  float* xe = x + (size_t)N_NODES * 512;

  // ---- carve workspace: f32 region then f16 region ----
  float* p = (float*)d_ws;
  auto carve = [&](size_t n) { float* r = p; p += n; return r; };
  float*    q     = carve((size_t)N_NODES * 128);
  float*    k     = carve((size_t)N_NODES * 128);
  float*    val   = carve((size_t)N_NODES * 128);
  float*    pq    = carve((size_t)3 * N_NODES * 64);
  float*    pk    = carve((size_t)3 * N_NODES * 64);
  float*    pv    = carve((size_t)3 * N_NODES * 64);
  float*    logit = carve((size_t)NE * 8);
  unsigned* mbuf  = (unsigned*)carve((size_t)N_NODES * 8);
  float*    denom = carve((size_t)N_NODES * 8);
  float*    out_s = carve((size_t)N_NODES * 128);
  float*    out_p = carve((size_t)3 * N_NODES * 64);
  float*    s_out = carve((size_t)N_NODES * 128);
  float*    v_out = carve((size_t)3 * N_NODES * 128);
  float*    eh1f  = carve((size_t)NE * 128);        // f32 partial accum
  float*    h2    = carve((size_t)N_NODES * 512);

  _Float16* ph = (_Float16*)p;
  auto carveh = [&](size_t n) { _Float16* r = ph; ph += n; return r; };
  _Float16* st_h    = carveh((size_t)N_NODES * 256);
  _Float16* sbuf_h  = carveh((size_t)N_NODES * 128);
  _Float16* vbuf_h  = carveh((size_t)3 * N_NODES * 128);
  _Float16* catb_h  = carveh((size_t)N_NODES * 192);
  _Float16* outp_h  = carveh((size_t)3 * N_NODES * 64);
  _Float16* eh1_h   = carveh((size_t)NE * 128);
  _Float16* xe_h    = carveh((size_t)NE * 128);
  _Float16* z_h     = carveh((size_t)N_NODES * 640);
  _Float16* h1_h    = carveh((size_t)N_NODES * 640);
  // transposed f16 weights, per layer
  _Float16* WqT  = carveh((size_t)NL * 256 * 128);
  _Float16* WkT  = carveh((size_t)NL * 256 * 128);
  _Float16* WvT  = carveh((size_t)NL * 256 * 128);
  _Float16* WpqT = carveh((size_t)NL * 128 * 64);
  _Float16* WpkT = carveh((size_t)NL * 128 * 64);
  _Float16* WpvT = carveh((size_t)NL * 128 * 64);
  _Float16* WosT = carveh((size_t)NL * 192 * 128);
  _Float16* WovT = carveh((size_t)NL * 64 * 128);
  _Float16* We1T = carveh((size_t)NL * 384 * 128);
  _Float16* We2T = carveh((size_t)NL * 128 * 128);
  _Float16* lW1T = carveh((size_t)NL * 640 * 640);
  _Float16* lW2T = carveh((size_t)NL * 640 * 512);

  hipMemcpyAsync(x,  in_x,  (size_t)N_NODES * 512 * sizeof(float),
                 hipMemcpyDeviceToDevice, stream);
  hipMemcpyAsync(xe, in_xe, (size_t)NE * 128 * sizeof(float),
                 hipMemcpyDeviceToDevice, stream);
  conv_h<<<nblk((size_t)NE * 128), 256, 0, stream>>>(in_xe, xe_h, NE * 128);

  // ---- one-time weight transpose/convert (tiny, L2-resident afterwards) ----
  for (int l = 0; l < NL; ++l) {
    wconv_t<<<nblk(256 * 128), 256, 0, stream>>>(Wq  + (size_t)l * 256 * 128, WqT  + (size_t)l * 256 * 128, 256, 128);
    wconv_t<<<nblk(256 * 128), 256, 0, stream>>>(Wk  + (size_t)l * 256 * 128, WkT  + (size_t)l * 256 * 128, 256, 128);
    wconv_t<<<nblk(256 * 128), 256, 0, stream>>>(Wv  + (size_t)l * 256 * 128, WvT  + (size_t)l * 256 * 128, 256, 128);
    wconv_t<<<nblk(128 * 64),  256, 0, stream>>>(Wpq + (size_t)l * 128 * 64,  WpqT + (size_t)l * 128 * 64,  128, 64);
    wconv_t<<<nblk(128 * 64),  256, 0, stream>>>(Wpk + (size_t)l * 128 * 64,  WpkT + (size_t)l * 128 * 64,  128, 64);
    wconv_t<<<nblk(128 * 64),  256, 0, stream>>>(Wpv + (size_t)l * 128 * 64,  WpvT + (size_t)l * 128 * 64,  128, 64);
    wconv_t<<<nblk(192 * 128), 256, 0, stream>>>(Wos + (size_t)l * 192 * 128, WosT + (size_t)l * 192 * 128, 192, 128);
    wconv_t<<<nblk(64 * 128),  256, 0, stream>>>(Wov + (size_t)l * 64 * 128,  WovT + (size_t)l * 64 * 128,  64, 128);
    wconv_t<<<nblk(384 * 128), 256, 0, stream>>>(We1 + (size_t)l * 384 * 128, We1T + (size_t)l * 384 * 128, 384, 128);
    wconv_t<<<nblk(128 * 128), 256, 0, stream>>>(We2 + (size_t)l * 128 * 128, We2T + (size_t)l * 128 * 128, 128, 128);
    wconv_t<<<nblk(640 * 640), 256, 0, stream>>>(lW1 + (size_t)l * 640 * 640, lW1T + (size_t)l * 640 * 640, 640, 640);
    wconv_t<<<nblk(640 * 512), 256, 0, stream>>>(lW2 + (size_t)l * 640 * 512, lW2T + (size_t)l * 640 * 512, 640, 512);
  }

  for (int l = 0; l < NL; ++l) {
    const float* We_l  = We  + (size_t)l * 128 * 8;
    const float* ga_l  = gam + (size_t)l * 8;
    const float* lb1_l = lb1 + (size_t)l * 640;
    const float* lb2_l = lb2 + (size_t)l * 512;
    const _Float16* WqT_l  = WqT  + (size_t)l * 256 * 128;
    const _Float16* WkT_l  = WkT  + (size_t)l * 256 * 128;
    const _Float16* WvT_l  = WvT  + (size_t)l * 256 * 128;
    const _Float16* WpqT_l = WpqT + (size_t)l * 128 * 64;
    const _Float16* WpkT_l = WpkT + (size_t)l * 128 * 64;
    const _Float16* WpvT_l = WpvT + (size_t)l * 128 * 64;
    const _Float16* WosT_l = WosT + (size_t)l * 192 * 128;
    const _Float16* WovT_l = WovT + (size_t)l * 64 * 128;
    const _Float16* We1T_l = We1T + (size_t)l * 384 * 128;  // Bt[n, k], ldbt=384
    const _Float16* We2T_l = We2T + (size_t)l * 128 * 128;
    const _Float16* lW1T_l = lW1T + (size_t)l * 640 * 640;
    const _Float16* lW2T_l = lW2T + (size_t)l * 640 * 512;

    // ---- projections ----
    build_st<<<nblk((size_t)N_NODES * 256), 256, 0, stream>>>(x, in_t, st_h, sbuf_h);
    pack_v<<<nblk((size_t)3 * N_NODES * 128), 256, 0, stream>>>(x, vbuf_h);
    gemm(stream, st_h, nullptr, 256, WqT_l, 256, q,   nullptr, 128, N_NODES, 256, 128, nullptr, 4);
    gemm(stream, st_h, nullptr, 256, WkT_l, 256, k,   nullptr, 128, N_NODES, 256, 128, nullptr, 4);
    gemm(stream, st_h, nullptr, 256, WvT_l, 256, val, nullptr, 128, N_NODES, 256, 128, nullptr, 4);
    gemm(stream, vbuf_h, nullptr, 128, WpqT_l, 128, pq, nullptr, 64, 3 * N_NODES, 128, 64, nullptr, 4);
    gemm(stream, vbuf_h, nullptr, 128, WpkT_l, 128, pk, nullptr, 64, 3 * N_NODES, 128, 64, nullptr, 4);
    gemm(stream, vbuf_h, nullptr, 128, WpvT_l, 128, pv, nullptr, 64, 3 * N_NODES, 128, 64, nullptr, 4);

    // ---- edge attention (segment softmax over dst) ----
    fill_u32<<<nblk((size_t)N_NODES * 8), 256, 0, stream>>>(mbuf, N_NODES * 8, 0u);
    fill_f32<<<nblk((size_t)N_NODES * 8), 256, 0, stream>>>(denom, N_NODES * 8, 0.f);
    fill_f32<<<nblk((size_t)N_NODES * 128), 256, 0, stream>>>(out_s, N_NODES * 128, 0.f);
    fill_f32<<<nblk((size_t)3 * N_NODES * 64), 256, 0, stream>>>(out_p, 3 * N_NODES * 64, 0.f);
    edge_logits<<<nblk((size_t)NE * 8), 256, 0, stream>>>(q, k, pq, pk, xe, We_l, ga_l,
                                                          src, dst, logit);
    seg_max<<<nblk((size_t)NE * 8), 256, 0, stream>>>(logit, dst, mbuf);
    edge_w<<<nblk((size_t)NE * 8), 256, 0, stream>>>(logit, dst, mbuf, denom);
    edge_agg<<<nblk((size_t)NE * 8), 256, 0, stream>>>(logit, src, dst, denom, val, pv,
                                                       out_s, out_p);

    // ---- node output projections + residual ----
    build_cat<<<nblk((size_t)N_NODES * 192), 256, 0, stream>>>(out_s, out_p, catb_h);
    conv_h<<<nblk((size_t)3 * N_NODES * 64), 256, 0, stream>>>(out_p, outp_h,
                                                               3 * N_NODES * 64);
    gemm(stream, catb_h, nullptr, 192, WosT_l, 192, s_out, nullptr, 128, N_NODES, 192, 128, nullptr, 4);
    gemm(stream, outp_h, nullptr, 64,  WovT_l, 64,  v_out, nullptr, 128, 3 * N_NODES, 64, 128, nullptr, 4);
    x_update1<<<nblk((size_t)N_NODES * 128), 256, 0, stream>>>(x, s_out, v_out);

    // ---- edge MLP: e_in@We1 as 3 K-slices of transposed We1, gelu on last ----
    gemm(stream, sbuf_h, src, 128, We1T_l,       384, eh1f, nullptr, 128, NE, 128, 128, nullptr, 4);
    gemm(stream, sbuf_h, dst, 128, We1T_l + 128, 384, eh1f, nullptr, 128, NE, 128, 128, nullptr, 5);
    gemm(stream, xe_h, nullptr, 128, We1T_l + 256, 384, eh1f, eh1_h, 128, NE, 128, 128, nullptr, 3);
    // xe += eh1 @ We2 ; also refresh xe_h for next layer
    gemm(stream, eh1_h, nullptr, 128, We2T_l, 128, xe, xe_h, 128, NE, 128, 128, nullptr, 5);

    // ---- local (rotation-frame) MLP ----
    build_z<<<nblk((size_t)N_NODES * 128), 256, 0, stream>>>(x, in_R, z_h);
    gemm(stream, z_h,  nullptr, 640, lW1T_l, 640, nullptr, h1_h, 640, N_NODES, 640, 640, lb1_l, 2);
    gemm(stream, h1_h, nullptr, 640, lW2T_l, 640, h2, nullptr, 512, N_NODES, 640, 512, lb2_l, 4);
    x_update2<<<nblk((size_t)N_NODES * 128), 256, 0, stream>>>(x, in_R, h2);
  }
}
